// DifferentiableScratchPad_75488345195291
// MI455X (gfx1250) — compile-verified, matrix operand order checked by reading, not run on hardware
//
#include <hip/hip_runtime.h>
#include <hip/hip_bf16.h>

// ---------------------------------------------------------------------------
// DifferentiableScratchPad for MI455X (gfx1250, wave32, WMMA bf16 path)
// N=65536, D=1024, S=512, K=64. Big GEMMs: 128x128 tiles, 8 waves, each wave
// owns a 32x64 sub-tile (2 A-frags x 4 B-frags -> 8 wmma per K-step),
// double-buffered LDS staging with b128 copies of pre-converted bf16 operands.
// ---------------------------------------------------------------------------

typedef unsigned short ushort_t;
typedef __attribute__((ext_vector_type(16))) __bf16 v16bf;
typedef __attribute__((ext_vector_type(8)))  float  v8f;

#define LP   34    // LDS pitch for legacy 64-tile f32 staging
#define LP2  66    // LDS pitch for 64-wide attn tiles
#define PB   40    // LDS pitch (ushorts) for b128-staged tiles: 80B rows, 16B aligned
#define NT   128

static constexpr int    NN = 65536;
static constexpr int    DD = 1024;
static constexpr int    SS = 512;
static constexpr int    KK = 64;
static constexpr float  SCALE = 0.04419417382415922f; // 1/sqrt(512)

// ---------------- helpers ----------------

__device__ __forceinline__ ushort_t f2bf(float f) {
  unsigned u = __float_as_uint(f);
  unsigned r = (u + 0x7FFFu + ((u >> 16) & 1u)) >> 16; // round-to-nearest-even
  return (ushort_t)r;
}

__device__ __forceinline__ v8f zero8() {
  v8f z;
#pragma unroll
  for (int i = 0; i < 8; i++) z[i] = 0.f;
  return z;
}

// Fragment gather per CDNA5 ISA 7.12.2 (16-bit 16x32 layout).
__device__ __forceinline__ v16bf load_frag(const ushort_t* base, int pitch) {
  int lane = threadIdx.x & 31;
  const ushort_t* p = base + (lane & 15) * pitch + ((lane & 16) ? 8 : 0);
  union { v16bf v; unsigned u[8]; } f;
#pragma unroll
  for (int j = 0; j < 4; j++) {
    f.u[j]     = *(const unsigned*)(p + 2 * j);
    f.u[4 + j] = *(const unsigned*)(p + 16 + 2 * j);
  }
  return f.v;
}

__device__ __forceinline__ v8f do_wmma(v16bf a, v16bf b, v8f c) {
  return __builtin_amdgcn_wmma_f32_16x16x32_bf16(false, a, false, b, (short)0,
                                                 c, false, false);
}

__device__ __forceinline__ void mma64(const ushort_t* Al, const ushort_t* Bl,
                                      int pitch, v8f acc[4]) {
  int w = threadIdx.x >> 5;
  v16bf a = load_frag(Al + w * 16 * pitch, pitch);
#pragma unroll
  for (int t = 0; t < 4; t++) {
    v16bf b = load_frag(Bl + t * 16 * pitch, pitch);
    acc[t] = do_wmma(a, b, acc[t]);
  }
}

// 128x128 tile, 8 waves: wave w -> rows (w&3)*32..+31, cols (w>>2)*64..+63.
// 6 fragment loads feed 8 wmma per K-step.
__device__ __forceinline__ void mma128x2(const ushort_t* Al, const ushort_t* Bl,
                                         v8f acc[2][4]) {
  int w = threadIdx.x >> 5;
  int rbase = (w & 3) * 32, cbase = (w >> 2) * 64;
  v16bf a0 = load_frag(Al + rbase * PB, PB);
  v16bf a1 = load_frag(Al + (rbase + 16) * PB, PB);
#pragma unroll
  for (int t = 0; t < 4; t++) {
    v16bf b = load_frag(Bl + (cbase + t * 16) * PB, PB);
    acc[0][t] = do_wmma(a0, b, acc[0][t]);
    acc[1][t] = do_wmma(a1, b, acc[1][t]);
  }
}

// ---- staging: b128 copies of bf16 tiles (rows x 32 k) into pitch-PB LDS ----

__device__ __forceinline__ void stage_bf16_128(const ushort_t* src, int ld,
                                               int r0, int k0, ushort_t* dst) {
  for (int i = threadIdx.x; i < 128 * 4; i += 256) {
    int r = i >> 2, c8 = (i & 3) << 3;
    *(uint4*)&dst[r * PB + c8] =
        *(const uint4*)&src[(size_t)(r0 + r) * ld + k0 + c8];
  }
}
__device__ __forceinline__ void stage_bf16_64(const ushort_t* src, int ld,
                                              int r0, int k0, ushort_t* dst) {
  for (int i = threadIdx.x; i < 64 * 4; i += NT) {
    int r = i >> 2, c8 = (i & 3) << 3;
    *(uint4*)&dst[r * PB + c8] =
        *(const uint4*)&src[(size_t)(r0 + r) * ld + k0 + c8];
  }
}

// legacy f32 staging (tiny state projections only)
__device__ __forceinline__ void stageA_f32(const float* A, int lda, int row0,
                                           int k0, ushort_t* Al) {
  for (int i = threadIdx.x; i < 64 * 32; i += NT) {
    int r = i >> 5, c = i & 31;
    Al[r * LP + c] = f2bf(A[(size_t)(row0 + r) * lda + (k0 + c)]);
  }
}
__device__ __forceinline__ void stageB_rm_f32(const float* B, int ldb, int k0,
                                              int n0, ushort_t* Bl) {
  for (int i = threadIdx.x; i < 64 * 32; i += NT) {
    int n = i & 63, k = i >> 6;
    Bl[n * LP + k] = f2bf(B[(size_t)(k0 + k) * ldb + (n0 + n)]);
  }
}
// transposed staging for aggregation B (bf16 row-major [K,N] -> Bt[n][k])
__device__ __forceinline__ void stageB_tr64_bf16(const ushort_t* B, int ldb,
                                                 int k0, int n0, ushort_t* Bl) {
  for (int i = threadIdx.x; i < 64 * 32; i += NT) {
    int n = i & 63, k = i >> 6;
    Bl[n * PB + k] = B[(size_t)(k0 + k) * ldb + (n0 + n)];
  }
}

// row-wise softmax over 64 cols held as 4 C-tiles (xor masks 1..8 stay per-row)
__device__ __forceinline__ void softmax64(v8f acc[4]) {
#pragma unroll
  for (int e = 0; e < 8; e++) {
    float m = fmaxf(fmaxf(acc[0][e], acc[1][e]), fmaxf(acc[2][e], acc[3][e]));
#pragma unroll
    for (int s = 1; s < 16; s <<= 1) m = fmaxf(m, __shfl_xor(m, s, 32));
    float p0 = __expf(acc[0][e] - m);
    float p1 = __expf(acc[1][e] - m);
    float p2 = __expf(acc[2][e] - m);
    float p3 = __expf(acc[3][e] - m);
    float sum = p0 + p1 + p2 + p3;
#pragma unroll
    for (int s = 1; s < 16; s <<= 1) sum += __shfl_xor(sum, s, 32);
    float inv = 1.f / sum;
    acc[0][e] = p0 * inv; acc[1][e] = p1 * inv;
    acc[2][e] = p2 * inv; acc[3][e] = p3 * inv;
  }
}

__device__ __forceinline__ float wred32(float v) {
#pragma unroll
  for (int s = 16; s > 0; s >>= 1) v += __shfl_xor(v, s, 32);
  return v;
}

// ---------------- conversion kernels ----------------

__global__ __launch_bounds__(256) void k_cvt(const float* __restrict__ in,
                                             ushort_t* __restrict__ out, size_t n) {
  size_t i = ((size_t)blockIdx.x * 256 + threadIdx.x) * 4;
  if (i < n) {
    float4 v = *(const float4*)(in + i);
    ushort_t o[4] = {f2bf(v.x), f2bf(v.y), f2bf(v.z), f2bf(v.w)};
    *(uint2*)(out + i) = *(const uint2*)o;
  }
}
// out[n*Kd + k] = bf16(in[k*Nd + n]) : weight [Kd,Nd] -> bf16 [Nd,Kd]
__global__ __launch_bounds__(256) void k_cvt_tr(const float* __restrict__ in,
                                                ushort_t* __restrict__ out,
                                                int Kd, int Nd) {
  int idx = blockIdx.x * 256 + threadIdx.x;
  if (idx < Kd * Nd) {
    int n = idx / Kd, k = idx % Kd;
    out[idx] = f2bf(in[(size_t)k * Nd + n]);
  }
}

// ---------------- kernels ----------------

// tiny 64-tile f32 GEMM (state projections): C_bf16 = A_f32 @ B_f32
__global__ __launch_bounds__(NT) void k_gemm_f32(
    const float* __restrict__ A, int lda, const float* __restrict__ B, int ldb,
    ushort_t* __restrict__ C, int ldc, int ksteps) {
  __shared__ __align__(16) ushort_t Al[64 * LP];
  __shared__ __align__(16) ushort_t Bl[64 * LP];
  int row0 = blockIdx.y * 64, n0 = blockIdx.x * 64;
  v8f acc[4];
#pragma unroll
  for (int t = 0; t < 4; t++) acc[t] = zero8();
  for (int ks = 0; ks < ksteps; ks++) {
    stageA_f32(A, lda, row0, ks * 32, Al);
    stageB_rm_f32(B, ldb, ks * 32, n0, Bl);
    __syncthreads();
    mma64(Al, Bl, LP, acc);
    __syncthreads();
  }
  int lane = threadIdx.x & 31, w = threadIdx.x >> 5;
  int half = (lane & 16) ? 8 : 0, cn = lane & 15;
#pragma unroll
  for (int t = 0; t < 4; t++)
#pragma unroll
    for (int e = 0; e < 8; e++)
      C[(size_t)(row0 + w * 16 + e + half) * ldc + n0 + t * 16 + cn] =
          f2bf(acc[t][e]);
}

// 128x128 bf16 GEMM, double-buffered. A [M,K] bf16, Bt [N,K] bf16.
// mode 0: bf16 out (*scale); 1: sigmoid(x+bias) bf16; 2: tanh(x+bias) bf16;
// mode 3: f32 out + residual.
__global__ __launch_bounds__(256) void k_gemm_bb(
    const ushort_t* __restrict__ A, int lda, const ushort_t* __restrict__ Bt,
    int ldb, void* __restrict__ Cout, int ldc, int ksteps, float scale,
    const float* __restrict__ bias, const float* __restrict__ resid, int mode) {
  __shared__ __align__(16) ushort_t Al[2][128 * PB];
  __shared__ __align__(16) ushort_t Bl[2][128 * PB];
  int row0 = blockIdx.y * 128, n0 = blockIdx.x * 128;
  v8f acc[2][4];
#pragma unroll
  for (int s = 0; s < 2; s++)
#pragma unroll
    for (int t = 0; t < 4; t++) acc[s][t] = zero8();

  stage_bf16_128(A, lda, row0, 0, Al[0]);
  stage_bf16_128(Bt, ldb, n0, 0, Bl[0]);
  __syncthreads();
  for (int ks = 0; ks < ksteps; ks++) {
    int cur = ks & 1, nxt = cur ^ 1;
    if (ks + 1 < ksteps) {
      stage_bf16_128(A, lda, row0, (ks + 1) * 32, Al[nxt]);
      stage_bf16_128(Bt, ldb, n0, (ks + 1) * 32, Bl[nxt]);
      if (ks + 2 < ksteps) {  // prefetch K-step+2 rows (global_prefetch_b8)
        int r = threadIdx.x >> 1;
        __builtin_prefetch(&A[(size_t)(row0 + r) * lda + (ks + 2) * 32], 0, 3);
        __builtin_prefetch(&Bt[(size_t)(n0 + r) * ldb + (ks + 2) * 32], 0, 3);
      }
    }
    mma128x2(Al[cur], Bl[cur], acc);
    __syncthreads();
  }

  int lane = threadIdx.x & 31, w = threadIdx.x >> 5;
  int half = (lane & 16) ? 8 : 0, cn = lane & 15;
  int rb = row0 + (w & 3) * 32, cb = n0 + (w >> 2) * 64;
  if (mode == 3) {
    float* H = (float*)Cout;
#pragma unroll
    for (int s = 0; s < 2; s++)
#pragma unroll
      for (int t = 0; t < 4; t++)
#pragma unroll
        for (int e = 0; e < 8; e++) {
          size_t r = (size_t)(rb + s * 16 + e + half);
          int    c = cb + t * 16 + cn;
          H[r * ldc + c] = acc[s][t][e] + resid[r * ldc + c];
        }
  } else {
    ushort_t* O = (ushort_t*)Cout;
#pragma unroll
    for (int s = 0; s < 2; s++)
#pragma unroll
      for (int t = 0; t < 4; t++)
#pragma unroll
        for (int e = 0; e < 8; e++) {
          size_t r = (size_t)(rb + s * 16 + e + half);
          int    c = cb + t * 16 + cn;
          float v = acc[s][t][e] * scale;
          if (mode == 1)      v = 1.f / (1.f + __expf(-(v + bias[c])));
          else if (mode == 2) v = tanhf(v + bias[c]);
          O[r * ldc + c] = f2bf(v);
        }
  }
}

// Fused attention: softmax(Q@Kp^T) then @ V[:,nch:nch+64]
__global__ __launch_bounds__(NT) void k_attn_av(
    const ushort_t* __restrict__ Q, const ushort_t* __restrict__ Kp,
    const ushort_t* __restrict__ V, ushort_t* __restrict__ AV) {
  __shared__ __align__(16) ushort_t Al[2][64 * PB];
  __shared__ __align__(16) ushort_t Bl[2][64 * PB];
  __shared__ __align__(16) ushort_t At[64 * LP2];
  __shared__ __align__(16) ushort_t Vt[64 * LP2];
  int row0 = blockIdx.y * 64, nch = blockIdx.x * 64;
  v8f acc[4];
#pragma unroll
  for (int t = 0; t < 4; t++) acc[t] = zero8();

  stage_bf16_64(Q, SS, row0, 0, Al[0]);
  stage_bf16_64(Kp, SS, 0, 0, Bl[0]);
  __syncthreads();
  for (int ks = 0; ks < 16; ks++) {              // K = 512
    int cur = ks & 1, nxt = cur ^ 1;
    if (ks < 15) {
      stage_bf16_64(Q, SS, row0, (ks + 1) * 32, Al[nxt]);
      stage_bf16_64(Kp, SS, 0, (ks + 1) * 32, Bl[nxt]);
    }
    mma64(Al[cur], Bl[cur], PB, acc);
    __syncthreads();
  }
  softmax64(acc);
  int lane = threadIdx.x & 31, w = threadIdx.x >> 5;
  int half = (lane & 16) ? 8 : 0, cn = lane & 15;
#pragma unroll
  for (int t = 0; t < 4; t++)
#pragma unroll
    for (int e = 0; e < 8; e++)
      At[(w * 16 + e + half) * LP2 + t * 16 + cn] = f2bf(acc[t][e]);
  for (int i = threadIdx.x; i < 64 * 64; i += NT) {
    int n = i & 63, k = i >> 6;                  // Vt[scol][slot]
    Vt[n * LP2 + k] = V[(size_t)k * SS + nch + n];
  }
  __syncthreads();
  v8f acc2[4];
#pragma unroll
  for (int t = 0; t < 4; t++) acc2[t] = zero8();
#pragma unroll
  for (int ks = 0; ks < 2; ks++)                 // K = 64 slots
    mma64(At + ks * 32, Vt + ks * 32, LP2, acc2);
#pragma unroll
  for (int t = 0; t < 4; t++)
#pragma unroll
    for (int e = 0; e < 8; e++)
      AV[(size_t)(row0 + w * 16 + e + half) * SS + nch + t * 16 + cn] =
          f2bf(acc2[t][e]);
}

// In-place LayerNorm (wave per row) + bf16 h copy + gate = sigmoid(h.Wg + bg)
__global__ __launch_bounds__(256) void k_ln_gate(
    float* __restrict__ H, ushort_t* __restrict__ Hb,
    const float* __restrict__ gamma, const float* __restrict__ beta,
    const float* __restrict__ Wg, const float* __restrict__ bg,
    float* __restrict__ gate) {
  int w = threadIdx.x >> 5, lane = threadIdx.x & 31;
  size_t row = (size_t)blockIdx.x * 8 + w;
  float* p = H + row * DD;
  float4 x[8];
  float s = 0.f, s2 = 0.f;
#pragma unroll
  for (int j = 0; j < 8; j++) {
    x[j] = *(const float4*)(p + j * 128 + lane * 4);
    s  += x[j].x + x[j].y + x[j].z + x[j].w;
    s2 += x[j].x * x[j].x + x[j].y * x[j].y + x[j].z * x[j].z + x[j].w * x[j].w;
  }
  s = wred32(s); s2 = wred32(s2);
  float mu  = s * (1.f / 1024.f);
  float var = s2 * (1.f / 1024.f) - mu * mu;
  float rstd = rsqrtf(var + 1e-6f);
  float gd = 0.f;
#pragma unroll
  for (int j = 0; j < 8; j++) {
    int c = j * 128 + lane * 4;
    float4 g = *(const float4*)(gamma + c);
    float4 b = *(const float4*)(beta + c);
    float4 wg = *(const float4*)(Wg + c);
    float4 y;
    y.x = (x[j].x - mu) * rstd * g.x + b.x;
    y.y = (x[j].y - mu) * rstd * g.y + b.y;
    y.z = (x[j].z - mu) * rstd * g.z + b.z;
    y.w = (x[j].w - mu) * rstd * g.w + b.w;
    *(float4*)(p + c) = y;
    ushort_t o[4] = {f2bf(y.x), f2bf(y.y), f2bf(y.z), f2bf(y.w)};
    *(uint2*)(Hb + row * DD + c) = *(const uint2*)o;
    gd += y.x * wg.x + y.y * wg.y + y.z * wg.z + y.w * wg.w;
  }
  gd = wred32(gd);
  if (lane == 0) gate[row] = 1.f / (1.f + __expf(-(gd + bg[0])));
}

// weighted^T = gate * softmax(T @ state^T)
__global__ __launch_bounds__(NT) void k_addr(
    const ushort_t* __restrict__ T, const ushort_t* __restrict__ stateB,
    const float* __restrict__ gate, ushort_t* __restrict__ wT) {
  __shared__ __align__(16) ushort_t Al[2][64 * PB];
  __shared__ __align__(16) ushort_t Bl[2][64 * PB];
  int row0 = blockIdx.x * 64;
  v8f acc[4];
#pragma unroll
  for (int t = 0; t < 4; t++) acc[t] = zero8();

  stage_bf16_64(T, SS, row0, 0, Al[0]);
  stage_bf16_64(stateB, SS, 0, 0, Bl[0]);
  __syncthreads();
  for (int ks = 0; ks < 16; ks++) {              // K = 512
    int cur = ks & 1, nxt = cur ^ 1;
    if (ks < 15) {
      stage_bf16_64(T, SS, row0, (ks + 1) * 32, Al[nxt]);
      stage_bf16_64(stateB, SS, 0, (ks + 1) * 32, Bl[nxt]);
    }
    mma64(Al[cur], Bl[cur], PB, acc);
    __syncthreads();
  }
  softmax64(acc);
  int lane = threadIdx.x & 31, w = threadIdx.x >> 5;
  int half = (lane & 16) ? 8 : 0, cn = lane & 15;
#pragma unroll
  for (int e = 0; e < 8; e++) {
    int r = row0 + w * 16 + e + half;
    float g = gate[r];
#pragma unroll
    for (int t = 0; t < 4; t++)
      wT[(size_t)(t * 16 + cn) * NN + r] = f2bf(acc[t][e] * g);
  }
}

// split-K aggregation: agg[64,512] += wT[64,krange] @ Mat[krange,512]
__global__ __launch_bounds__(NT) void k_agg(
    const ushort_t* __restrict__ wT, const ushort_t* __restrict__ Mat,
    float* __restrict__ agg, int kRowsPerSplit) {
  __shared__ __align__(16) ushort_t Al[2][64 * PB];
  __shared__ __align__(16) ushort_t Bl[2][64 * PB];
  int n0 = blockIdx.x * 64;
  int k0base = blockIdx.y * kRowsPerSplit;
  int nsteps = kRowsPerSplit / 32;
  v8f acc[4];
#pragma unroll
  for (int t = 0; t < 4; t++) acc[t] = zero8();

  stage_bf16_64(wT, NN, 0, k0base, Al[0]);
  stageB_tr64_bf16(Mat, SS, k0base, n0, Bl[0]);
  __syncthreads();
  for (int ks = 0; ks < nsteps; ks++) {
    int cur = ks & 1, nxt = cur ^ 1;
    if (ks + 1 < nsteps) {
      int k0 = k0base + (ks + 1) * 32;
      stage_bf16_64(wT, NN, 0, k0, Al[nxt]);
      stageB_tr64_bf16(Mat, SS, k0, n0, Bl[nxt]);
    }
    mma64(Al[cur], Bl[cur], PB, acc);
    __syncthreads();
  }
  int lane = threadIdx.x & 31, w = threadIdx.x >> 5;
  int half = (lane & 16) ? 8 : 0, cn = lane & 15;
#pragma unroll
  for (int t = 0; t < 4; t++)
#pragma unroll
    for (int e = 0; e < 8; e++)
      atomicAdd(&agg[(size_t)(w * 16 + e + half) * SS + n0 + t * 16 + cn],
                acc[t][e]);
}

__global__ void k_zero(float* p, int n) {
  int i = blockIdx.x * blockDim.x + threadIdx.x;
  if (i < n) p[i] = 0.f;
}

__global__ void k_new_state(const float* __restrict__ state,
                            const float* __restrict__ er,
                            const float* __restrict__ wr,
                            float* __restrict__ out) {
  int i = blockIdx.x * blockDim.x + threadIdx.x;
  if (i < KK * SS) {
    float e = fminf(fmaxf(er[i], 0.f), 1.f);
    out[i] = state[i] * (1.f - e) + wr[i];
  }
}

// ---------------- launch ----------------

extern "C" void kernel_launch(void* const* d_in, const int* in_sizes, int n_in,
                              void* d_out, int out_size, void* d_ws, size_t ws_size,
                              hipStream_t stream) {
  (void)in_sizes; (void)n_in; (void)out_size; (void)ws_size;
  const float* X     = (const float*)d_in[0];
  const float* state = (const float*)d_in[1];
  const float* Wq    = (const float*)d_in[2];
  const float* Wk    = (const float*)d_in[3];
  const float* Wv    = (const float*)d_in[4];
  const float* Wo    = (const float*)d_in[5];
  const float* gam   = (const float*)d_in[6];
  const float* bet   = (const float*)d_in[7];
  const float* Wa    = (const float*)d_in[8];
  const float* Wg    = (const float*)d_in[9];
  const float* bg    = (const float*)d_in[10];
  const float* We    = (const float*)d_in[11];
  const float* be    = (const float*)d_in[12];
  const float* Wc    = (const float*)d_in[13];
  const float* bc    = (const float*)d_in[14];

  float* Hout = (float*)d_out;
  float* Sout = (float*)d_out + (size_t)NN * DD;

  char* ws = (char*)d_ws;
  ushort_t* KpB    = (ushort_t*)(ws + 0);                   // 64KB
  ushort_t* VB     = (ushort_t*)(ws + 65536);               // 64KB
  float*    gate   = (float*)   (ws + 131072);              // 256KB
  ushort_t* wT     = (ushort_t*)(ws + 393216);              // 8MB [64,N]
  float*    erAgg  = (float*)   (ws + 8781824);             // 128KB
  float*    wrAgg  = (float*)   (ws + 8912896);             // 128KB
  ushort_t* WqT    = (ushort_t*)(ws + 9437184);             // [512,1024] 1MB
  ushort_t* WoT    = (ushort_t*)(ws + 10485760);            // [1024,512] 1MB
  ushort_t* WaT    = (ushort_t*)(ws + 11534336);            // [512,1024] 1MB
  ushort_t* WeT    = (ushort_t*)(ws + 12582912);            // [512,1024] 1MB
  ushort_t* WcT    = (ushort_t*)(ws + 13631488);            // [512,1024] 1MB
  ushort_t* stateB = (ushort_t*)(ws + 14680064);            // 64KB
  ushort_t* buf0   = (ushort_t*)(ws + (size_t)16777216);    // 64MB (Q->T->E)
  ushort_t* buf1   = (ushort_t*)(ws + (size_t)83886080);    // 64MB (AV->C)
  ushort_t* XHb    = (ushort_t*)(ws + (size_t)150994944);   // 128MB (Xb then Hb)

  // 0. one-time conversions (weights transposed to [N,K] bf16; X, state to bf16)
  k_cvt<<<(size_t)NN * DD / 1024, 256, 0, stream>>>(X, XHb, (size_t)NN * DD);
  k_cvt<<<KK * SS / 1024, 256, 0, stream>>>(state, stateB, (size_t)KK * SS);
  k_cvt_tr<<<DD * SS / 256, 256, 0, stream>>>(Wq, WqT, DD, SS);
  k_cvt_tr<<<SS * DD / 256, 256, 0, stream>>>(Wo, WoT, SS, DD);
  k_cvt_tr<<<DD * SS / 256, 256, 0, stream>>>(Wa, WaT, DD, SS);
  k_cvt_tr<<<DD * SS / 256, 256, 0, stream>>>(We, WeT, DD, SS);
  k_cvt_tr<<<DD * SS / 256, 256, 0, stream>>>(Wc, WcT, DD, SS);
  // 1. slot projections: Kp = state@Wk, V = state@Wv (f32 path, tiny)
  k_gemm_f32<<<dim3(8, 1), NT, 0, stream>>>(state, SS, Wk, SS, KpB, SS, 16);
  k_gemm_f32<<<dim3(8, 1), NT, 0, stream>>>(state, SS, Wv, SS, VB,  SS, 16);
  // 2. Q = X@Wq * 1/sqrt(S)
  k_gemm_bb<<<dim3(4, 512), 256, 0, stream>>>(XHb, DD, WqT, DD, buf0, SS, 32,
                                              SCALE, nullptr, nullptr, 0);
  // 3. attn + attn@V
  k_attn_av<<<dim3(8, 1024), NT, 0, stream>>>(buf0, KpB, VB, buf1);
  // 4. h_pre = AV@Wo + X  (f32 into d_out)
  k_gemm_bb<<<dim3(8, 512), 256, 0, stream>>>(buf1, SS, WoT, SS, Hout, DD, 16,
                                              1.f, nullptr, X, 3);
  // 5. LayerNorm in-place + bf16 h + gate
  k_ln_gate<<<NN / 8, 256, 0, stream>>>(Hout, XHb, gam, bet, Wg, bg, gate);
  // 6. T = h@Wa ; weighted^T = gate * softmax(T@state^T)
  k_gemm_bb<<<dim3(4, 512), 256, 0, stream>>>(XHb, DD, WaT, DD, buf0, SS, 32,
                                              1.f, nullptr, nullptr, 0);
  k_addr<<<dim3(1024), NT, 0, stream>>>(buf0, stateB, gate, wT);
  // 7. E = sigmoid(h@We+be), C = tanh(h@Wc+bc)
  k_gemm_bb<<<dim3(4, 512), 256, 0, stream>>>(XHb, DD, WeT, DD, buf0, SS, 32,
                                              1.f, be, nullptr, 1);
  k_gemm_bb<<<dim3(4, 512), 256, 0, stream>>>(XHb, DD, WcT, DD, buf1, SS, 32,
                                              1.f, bc, nullptr, 2);
  // 8. aggregation (split-K, f32 atomics); erAgg/wrAgg contiguous -> zero both
  k_zero<<<(2 * KK * SS + 255) / 256, 256, 0, stream>>>(erAgg, 2 * KK * SS);
  k_agg<<<dim3(8, 32), NT, 0, stream>>>(wT, buf0, erAgg, NN / 32);
  k_agg<<<dim3(8, 32), NT, 0, stream>>>(wT, buf1, wrAgg, NN / 32);
  // 9. new_state
  k_new_state<<<(KK * SS + 255) / 256, 256, 0, stream>>>(state, erAgg, wrAgg, Sout);
}
